// Top2Gating_42296837931656
// MI455X (gfx1250) — matrix operand order repaired and checked by reading, not compiled, not run on hardware
//
#include <hip/hip_runtime.h>
#include <hip/hip_bf16.h>
#include <math.h>

// Problem constants (reference: B,N,D,E = 4,4096,1024,64)
#define B_    4
#define N_    4096
#define D_    1024
#define E_    64
#define CAP_  128                 // expert capacity = max(min(N, N*2/E), 4)
#define T_    (B_ * N_)           // 16384 tokens
#define THRESH_ 0.2f
#define EPS_    1e-9f

static const size_t BNEC = (size_t)T_ * E_ * CAP_;   // 134,217,728 elements per tensor

typedef __attribute__((ext_vector_type(16))) _Float16 v16h;
typedef __attribute__((ext_vector_type(8)))  float    v8f;

// ---------------------------------------------------------------------------
// Kernel 1: gate_logits[t, e] = sum_k x[t,k] * W[k,e]  via v_wmma_f32_16x16x32_f16
// One wave (32 lanes) owns a 16-token x 64-expert tile: 4 accumulators, K-loop of 32.
// ---------------------------------------------------------------------------
__global__ __launch_bounds__(256)
void gemm_logits_wmma(const float* __restrict__ x, const float* __restrict__ w,
                      float* __restrict__ logits) {
    const int lane = threadIdx.x & 31;
    const int wave = threadIdx.x >> 5;
    const int tile = blockIdx.x * 8 + wave;        // 16-token tile id
    const int t0   = tile * 16;
    const int m    = lane & 15;                    // row / column within tile
    const int h    = lane >> 4;                    // lane-half

    const float* __restrict__ xrow = x + (size_t)(t0 + m) * D_;

    v8f acc[4] = {};                               // 4 expert tiles of 16

    for (int k0 = 0; k0 < D_; k0 += 32) {
        // ---- A fragment: 16x32, contiguous k order (not AMD's table order). Lane-half h: VGPR0-3 hold
        // K=16h..16h+7 (packed pairs), VGPR4-7 hold K=16h+8..16h+15.
        const float* pa0 = xrow + k0 + 16 * h;      // contiguous 8 floats
        const float* pa1 = xrow + k0 + 16 * h + 8;  // contiguous 8 floats
        float4 f0 = ((const float4*)pa0)[0];
        float4 f1 = ((const float4*)pa0)[1];
        float4 f2 = ((const float4*)pa1)[0];
        float4 f3 = ((const float4*)pa1)[1];
        if (k0 + 32 < D_) __builtin_prefetch(xrow + k0 + 32, 0, 1);  // global_prefetch_b8

        v16h a;
        a[0]=(_Float16)f0.x; a[1]=(_Float16)f0.y; a[2]=(_Float16)f0.z; a[3]=(_Float16)f0.w;
        a[4]=(_Float16)f1.x; a[5]=(_Float16)f1.y; a[6]=(_Float16)f1.z; a[7]=(_Float16)f1.w;
        a[8]=(_Float16)f2.x; a[9]=(_Float16)f2.y; a[10]=(_Float16)f2.z; a[11]=(_Float16)f2.w;
        a[12]=(_Float16)f3.x; a[13]=(_Float16)f3.y; a[14]=(_Float16)f3.z; a[15]=(_Float16)f3.w;

        // ---- B fragments: 32x16, same contiguous k order as A. Lane-half h covers K=16h..16h+15,
        // column N = lane%16, VGPR v packs K=16h+2v, 16h+2v+1.
        #pragma unroll
        for (int et = 0; et < 4; ++et) {
            const float* pw = w + (size_t)(k0 + 16 * h) * E_ + et * 16 + m;
            v16h bf;
            #pragma unroll
            for (int j = 0; j < 16; ++j) bf[j] = (_Float16)pw[(size_t)j * E_];

            acc[et] = __builtin_amdgcn_wmma_f32_16x16x32_f16(
                /*neg_a=*/false, a, /*neg_b=*/false, bf,
                /*c_mod=*/(short)0, acc[et], /*reuse_a=*/false, /*reuse_b=*/false);
        }
    }

    // C/D layout: VGPR r -> token row (r + 8h), column = lane%16.
    #pragma unroll
    for (int et = 0; et < 4; ++et)
        #pragma unroll
        for (int r = 0; r < 8; ++r)
            logits[(size_t)(t0 + r + 8 * h) * E_ + et * 16 + m] = acc[et][r];
}

// ---------------------------------------------------------------------------
// Kernel 2: per-token softmax over E=64, top-1/top-2, normalize, threshold.
// Accumulates density_proxy[b,e], top1 counts[b,e], z-loss via LDS then global atomics.
// ---------------------------------------------------------------------------
__global__ __launch_bounds__(256)
void softmax_top2(const float* __restrict__ logits,
                  float* __restrict__ g1, float* __restrict__ g2,
                  int* __restrict__ idx1, int* __restrict__ idx2, int* __restrict__ keep2,
                  float* __restrict__ dproxy, float* __restrict__ cnt1,
                  float* __restrict__ zsum) {
    __shared__ float s_dp[E_];
    __shared__ float s_c1[E_];
    __shared__ float s_z;
    const int tid = threadIdx.x;
    if (tid < E_) { s_dp[tid] = 0.f; s_c1[tid] = 0.f; }
    if (tid == 0) s_z = 0.f;
    __syncthreads();

    const int t = blockIdx.x * 256 + tid;
    const int b = t / N_;
    const float* __restrict__ lg = logits + (size_t)t * E_;

    // pass 1: max + argmax (first occurrence) + second argmax
    float mx = -INFINITY; int i1 = 0;
    for (int e = 0; e < E_; ++e) { float v = lg[e]; if (v > mx) { mx = v; i1 = e; } }
    float m2 = -INFINITY; int i2 = 0;
    for (int e = 0; e < E_; ++e) { if (e == i1) continue; float v = lg[e]; if (v > m2) { m2 = v; i2 = e; } }

    // pass 2: softmax denominator
    float sum = 0.f;
    for (int e = 0; e < E_; ++e) sum += __expf(lg[e] - mx);
    const float inv = 1.0f / sum;
    const float lse = mx + __logf(sum);

    const float p1 = __expf(lg[i1] - mx) * inv;   // top-1 softmax prob
    const float p2 = __expf(m2    - mx) * inv;    // top-2 softmax prob
    const float denom = p1 + p2 + EPS_;
    const float g1n = p1 / denom;
    const float g2n = p2 / denom;

    g1[t] = g1n; g2[t] = g2n;
    idx1[t] = i1; idx2[t] = i2;
    keep2[t] = (g2n > THRESH_) ? 1 : 0;

    // pass 3: density_proxy accumulation
    for (int e = 0; e < E_; ++e) atomicAdd(&s_dp[e], __expf(lg[e] - mx) * inv);
    atomicAdd(&s_c1[i1], 1.0f);
    atomicAdd(&s_z, lse);
    __syncthreads();

    if (tid < E_) {
        atomicAdd(&dproxy[b * E_ + tid], s_dp[tid]);
        atomicAdd(&cnt1[b * E_ + tid],   s_c1[tid]);
    }
    if (tid == 0) atomicAdd(zsum, s_z);
}

// ---------------------------------------------------------------------------
// Kernel 3: ordered exclusive cumsum over tokens per (batch, expert).
// One block per batch; 256-token chunks processed in order; intra-chunk rank by scan.
// ---------------------------------------------------------------------------
__global__ __launch_bounds__(256)
void positions_kernel(const int* __restrict__ idx1, const int* __restrict__ idx2,
                      const int* __restrict__ keep2,
                      int* __restrict__ pos1, int* __restrict__ kept1,
                      int* __restrict__ pos2, int* __restrict__ kept2f) {
    __shared__ int runc[E_];
    __shared__ int cap1[E_];
    __shared__ int sidx[256];
    const int b = blockIdx.x;
    const int tid = threadIdx.x;

    if (tid < E_) runc[tid] = 0;
    __syncthreads();

    // ---- pass 1: top-1 positions (cumsum of mask_1, pre-capacity)
    for (int c = 0; c < N_ / 256; ++c) {
        const int t = b * N_ + c * 256 + tid;
        const int e = idx1[t];
        sidx[tid] = e;
        __syncthreads();
        int rank = 0;
        for (int j = 0; j < tid; ++j) rank += (sidx[j] == e);
        const int p = runc[e] + rank;
        pos1[t] = p;
        kept1[t] = (p < CAP_) ? 1 : 0;
        __syncthreads();
        atomicAdd(&runc[e], 1);
        __syncthreads();
    }
    if (tid < E_) cap1[tid] = (runc[tid] < CAP_) ? runc[tid] : CAP_;  // mask_1_count (capped)
    __syncthreads();
    if (tid < E_) runc[tid] = 0;
    __syncthreads();

    // ---- pass 2: top-2 positions (cumsum of thresholded mask_2, offset by mask_1_count)
    for (int c = 0; c < N_ / 256; ++c) {
        const int t = b * N_ + c * 256 + tid;
        const int e = idx2[t];
        const int k = keep2[t];
        sidx[tid] = k ? e : -1;
        __syncthreads();
        int rank = 0;
        for (int j = 0; j < tid; ++j) rank += (sidx[j] == e);
        const int p = cap1[e] + runc[e] + rank;
        pos2[t] = p;
        kept2f[t] = (k && p < CAP_) ? 1 : 0;
        __syncthreads();
        if (k) atomicAdd(&runc[e], 1);
        __syncthreads();
    }
}

// ---------------------------------------------------------------------------
// Kernel 4: bulk zero-fill of dispatch + combine tensors (2 * BNEC floats), float4 stores.
// ---------------------------------------------------------------------------
__global__ __launch_bounds__(256)
void zero_out(float4* __restrict__ out) {
    const size_t i = (size_t)blockIdx.x * 256 + threadIdx.x;
    out[i] = make_float4(0.f, 0.f, 0.f, 0.f);
}

__global__ void zero_accum(float* __restrict__ dproxy, float* __restrict__ cnt1,
                           float* __restrict__ zsum) {
    const int tid = threadIdx.x;
    if (tid < B_ * E_) { dproxy[tid] = 0.f; cnt1[tid] = 0.f; }
    if (tid == 0) zsum[0] = 0.f;
}

// ---------------------------------------------------------------------------
// Kernel 5: sparse scatter of gates into dispatch / combine tensors.
// ---------------------------------------------------------------------------
__global__ __launch_bounds__(256)
void scatter_kernel(const float* __restrict__ g1, const float* __restrict__ g2,
                    const int* __restrict__ idx1, const int* __restrict__ idx2,
                    const int* __restrict__ pos1, const int* __restrict__ kept1,
                    const int* __restrict__ pos2, const int* __restrict__ kept2f,
                    float* __restrict__ out) {
    const int t = blockIdx.x * 256 + threadIdx.x;
    const size_t combine_off = BNEC;
    if (kept1[t]) {
        const size_t o = ((size_t)t * E_ + idx1[t]) * CAP_ + pos1[t];
        out[o] = 1.0f;
        out[combine_off + o] = g1[t];
    }
    if (kept2f[t]) {
        const size_t o = ((size_t)t * E_ + idx2[t]) * CAP_ + pos2[t];
        out[o] = 1.0f;
        out[combine_off + o] = g2[t];
    }
}

// ---------------------------------------------------------------------------
// Kernel 6: scalar losses.
// loss = mean_{b,e}(density_proxy * density_1) * E^2 ;  z = mean_b sum_n lse
// ---------------------------------------------------------------------------
__global__ void finalize_kernel(const float* __restrict__ dproxy, const float* __restrict__ cnt1,
                                const float* __restrict__ zsum, float* __restrict__ out_scalars) {
    if (threadIdx.x == 0 && blockIdx.x == 0) {
        float acc = 0.f;
        for (int i = 0; i < B_ * E_; ++i)
            acc += (dproxy[i] / (float)N_) * (cnt1[i] / (float)N_);
        out_scalars[0] = acc / (float)(B_ * E_) * (float)(E_ * E_);
        out_scalars[1] = zsum[0] / (float)B_;
    }
}

// ---------------------------------------------------------------------------
extern "C" void kernel_launch(void* const* d_in, const int* in_sizes, int n_in,
                              void* d_out, int out_size, void* d_ws, size_t ws_size,
                              hipStream_t stream) {
    const float* x = (const float*)d_in[0];       // [B,N,D] fp32
    const float* w = (const float*)d_in[1];       // [D,E]   fp32
    float* out = (float*)d_out;                   // dispatch | combine | loss | z_loss

    // workspace carve-out
    char* ws = (char*)d_ws;
    float* logits = (float*)ws;            ws += (size_t)T_ * E_ * sizeof(float);
    float* g1     = (float*)ws;            ws += (size_t)T_ * sizeof(float);
    float* g2     = (float*)ws;            ws += (size_t)T_ * sizeof(float);
    int*   idx1   = (int*)ws;              ws += (size_t)T_ * sizeof(int);
    int*   idx2   = (int*)ws;              ws += (size_t)T_ * sizeof(int);
    int*   keep2  = (int*)ws;              ws += (size_t)T_ * sizeof(int);
    int*   pos1   = (int*)ws;              ws += (size_t)T_ * sizeof(int);
    int*   kept1  = (int*)ws;              ws += (size_t)T_ * sizeof(int);
    int*   pos2   = (int*)ws;              ws += (size_t)T_ * sizeof(int);
    int*   kept2f = (int*)ws;              ws += (size_t)T_ * sizeof(int);
    float* dproxy = (float*)ws;            ws += (size_t)B_ * E_ * sizeof(float);
    float* cnt1   = (float*)ws;            ws += (size_t)B_ * E_ * sizeof(float);
    float* zsum   = (float*)ws;            ws += sizeof(float);

    // 0) reset small accumulators (ws is poisoned, not re-zeroed between replays)
    zero_accum<<<1, 256, 0, stream>>>(dproxy, cnt1, zsum);

    // 1) WMMA GEMM: logits = x @ W   (128 blocks x 8 waves x 16-token tiles)
    gemm_logits_wmma<<<T_ / (16 * 8), 256, 0, stream>>>(x, w, logits);

    // 2) softmax / top-2 / threshold + loss accumulators
    softmax_top2<<<T_ / 256, 256, 0, stream>>>(logits, g1, g2, idx1, idx2, keep2,
                                               dproxy, cnt1, zsum);

    // 3) ordered capacity positions (one block per batch)
    positions_kernel<<<B_, 256, 0, stream>>>(idx1, idx2, keep2, pos1, kept1, pos2, kept2f);

    // 4) zero-fill the 1.07 GB dispatch+combine tensors (bandwidth-dominant step)
    const size_t nvec4 = (2 * BNEC) / 4;                    // 67,108,864 float4 stores
    zero_out<<<(unsigned)(nvec4 / 256), 256, 0, stream>>>((float4*)out);

    // 5) sparse scatter of gates
    scatter_kernel<<<T_ / 256, 256, 0, stream>>>(g1, g2, idx1, idx2, pos1, kept1,
                                                 pos2, kept2f, out);

    // 6) scalar losses at the tail of d_out
    finalize_kernel<<<1, 64, 0, stream>>>(dproxy, cnt1, zsum, out + 2 * BNEC);
}
